// AvgPoolingInitializer_64922725646383
// MI455X (gfx1250) — matrix operand order repaired and checked by reading, not compiled, not run on hardware
//
#include <hip/hip_runtime.h>
#include <hip/hip_bf16.h>

// Problem constants (from reference setup_inputs)
#define BB    8
#define CC    256
#define NSP   16384      // H*W = 128*128
#define NFG   16         // I
#define NBG   4          // Qb
#define NQ    20         // I + Qb
#define ROWS  32         // padded query rows (2 WMMA M-tiles)
#define KCH   32         // number of split-K chunks
#define KLEN  (NSP/KCH)  // 512 spatial positions per chunk

typedef __attribute__((ext_vector_type(16))) _Float16 v16h;
typedef __attribute__((ext_vector_type(8)))  float    v8f;
typedef __attribute__((ext_vector_type(4)))  float    vf4;
typedef __attribute__((ext_vector_type(4)))  unsigned int v4u;

union H16 { v4u u[2]; v16h h; };

// ---------------------------------------------------------------------------
// 1) Per-(b,q) mask stats: count(>0.5) and first-argmax over N=16384
// ---------------------------------------------------------------------------
__global__ __launch_bounds__(256)
void mask_stats_kernel(const float* __restrict__ fg, const float* __restrict__ bg,
                       int* __restrict__ cnt, int* __restrict__ amax) {
    int rid = blockIdx.x;                 // 0 .. B*NQ-1
    int b = rid / NQ, q = rid % NQ;
    const float* m = (q < NFG) ? fg + ((size_t)(b * NFG + q)) * NSP
                               : bg + ((size_t)(b * NBG + (q - NFG))) * NSP;
    int t = threadIdx.x;
    int lc = 0; float bv = -1.0f; int bi = 0;
    for (int n = t; n < NSP; n += 256) {
        float v = m[n];
        lc += (v > 0.5f) ? 1 : 0;
        if (v > bv) { bv = v; bi = n; }   // strict >: keeps first (lowest n) max
    }
    __shared__ int   sc[256];
    __shared__ float sv[256];
    __shared__ int   si[256];
    sc[t] = lc; sv[t] = bv; si[t] = bi;
    __syncthreads();
    for (int s = 128; s > 0; s >>= 1) {
        if (t < s) {
            sc[t] += sc[t + s];
            if (sv[t + s] > sv[t] || (sv[t + s] == sv[t] && si[t + s] < si[t])) {
                sv[t] = sv[t + s]; si[t] = si[t + s];
            }
        }
        __syncthreads();
    }
    if (t == 0) { cnt[b * ROWS + q] = sc[0]; amax[b * ROWS + q] = si[0]; }
}

// ---------------------------------------------------------------------------
// 2) Build padded f16 weight matrix W[b][32][N]; rows 20..31 are zero
// ---------------------------------------------------------------------------
__global__ __launch_bounds__(256)
void build_w_kernel(const float* __restrict__ fg, const float* __restrict__ bg,
                    const int* __restrict__ cnt, const int* __restrict__ amax,
                    _Float16* __restrict__ W) {
    size_t tid = (size_t)blockIdx.x * blockDim.x + threadIdx.x;
    size_t total = (size_t)BB * ROWS * NSP;
    if (tid >= total) return;
    int n   = (int)(tid % NSP);
    int row = (int)((tid / NSP) % ROWS);
    int b   = (int)(tid / ((size_t)NSP * ROWS));
    float w = 0.0f;
    if (row < NFG) {
        int c = cnt[b * ROWS + row];
        if (c > 0) {
            float v = fg[((size_t)(b * NFG + row)) * NSP + n];
            w = (v > 0.5f) ? 1.0f : 0.0f;
        } else {
            w = (n == amax[b * ROWS + row]) ? 1.0f : 0.0f;   // one-hot fallback
        }
    } else if (row < NQ) {
        float v = bg[((size_t)(b * NBG + (row - NFG))) * NSP + n];
        w = (v > 0.5f) ? 1.0f : 0.0f;
    }
    W[tid] = (_Float16)w;
}

// ---------------------------------------------------------------------------
// 3) Zero the f32 accumulator scratch [B][32][C]
// ---------------------------------------------------------------------------
__global__ __launch_bounds__(256)
void zero_acc_kernel(float* __restrict__ acc) {
    int t = blockIdx.x * blockDim.x + threadIdx.x;
    if (t < BB * ROWS * CC) acc[t] = 0.0f;
}

// ---------------------------------------------------------------------------
// 4) Split-K WMMA GEMM: acc[b][row][c] += sum_k W[b][row][k] * fmap[b][c][k]
//    1 wave per block; bid = b*512 + ctile*32 + kchunk
// ---------------------------------------------------------------------------
__global__ __launch_bounds__(32)
void wmma_gemm_kernel(const float* __restrict__ fmap, const _Float16* __restrict__ W,
                      float* __restrict__ acc) {
    int bid = blockIdx.x;
    int kc  = bid & (KCH - 1);
    int ct  = (bid >> 5) & 15;
    int b   = bid >> 9;
    int lane = threadIdx.x & 31;
    int lh   = lane >> 4;     // half-wave select
    int lm   = lane & 15;

    // B-matrix source: channel row (K-contiguous in memory)
    const float*    fr = fmap + ((size_t)b * CC + (size_t)(ct * 16 + lm)) * NSP;
    // A-matrix sources: query rows for M-tile 0 (rows 0-15) and tile 1 (16-31)
    const _Float16* w0 = W + ((size_t)b * ROWS + lm) * NSP;
    const _Float16* w1 = w0 + (size_t)16 * NSP;

    v8f acc0 = {};
    v8f acc1 = {};
    int kbase = kc * KLEN;

    for (int kk = 0; kk < KLEN; kk += 32) {
        int k0 = kbase + kk;

        // ---- B fragment: 32x16 f16; lanes0-15 hold K 0..15, lanes16-31 K 16..31
        int bofs = k0 + lh * 16;
        vf4 f0 = *(const vf4*)(fr + bofs);
        vf4 f1 = *(const vf4*)(fr + bofs + 4);
        vf4 f2 = *(const vf4*)(fr + bofs + 8);
        vf4 f3 = *(const vf4*)(fr + bofs + 12);
        __builtin_prefetch(fr + bofs + 32, 0, 1);   // next K-step (128B line)
        v16h bf;
#pragma unroll
        for (int j = 0; j < 4; ++j) {
            bf[j]      = (_Float16)f0[j];
            bf[4 + j]  = (_Float16)f1[j];
            bf[8 + j]  = (_Float16)f2[j];
            bf[12 + j] = (_Float16)f3[j];
        }

        // ---- A fragments: 16x32 f16; lanes0-15 hold K 0..15, lanes16-31 K 16..31
        int aofs = k0 + lh * 16;
        H16 a0h, a1h;
        a0h.u[0] = *(const v4u*)(w0 + aofs);
        a0h.u[1] = *(const v4u*)(w0 + aofs + 8);
        a1h.u[0] = *(const v4u*)(w1 + aofs);
        a1h.u[1] = *(const v4u*)(w1 + aofs + 8);

        acc0 = __builtin_amdgcn_wmma_f32_16x16x32_f16(false, a0h.h, false, bf,
                                                      (short)0, acc0, false, false);
        acc1 = __builtin_amdgcn_wmma_f32_16x16x32_f16(false, a1h.h, false, bf,
                                                      (short)0, acc1, false, false);
    }

    // C/D layout: VGPR r, lane l -> row = r + (l>=16 ? 8 : 0), col = l&15
    int col = ct * 16 + lm;
#pragma unroll
    for (int r = 0; r < 8; ++r) {
        int m0 = lh * 8 + r;
        atomicAdd(&acc[((size_t)b * ROWS + m0) * CC + col],      acc0[r]);
        atomicAdd(&acc[((size_t)b * ROWS + 16 + m0) * CC + col], acc1[r]);
    }
}

// ---------------------------------------------------------------------------
// 5) Finalize: divide by counts, apply fallbacks, write fg then bg to d_out
// ---------------------------------------------------------------------------
__global__ __launch_bounds__(256)
void finalize_kernel(const float* __restrict__ acc, const int* __restrict__ cnt,
                     float* __restrict__ out) {
    int tid = blockIdx.x * blockDim.x + threadIdx.x;   // B * NQ * C
    if (tid >= BB * NQ * CC) return;
    int c = tid % CC;
    int q = (tid / CC) % NQ;
    int b = tid / (CC * NQ);
    float s  = acc[((size_t)b * ROWS + q) * CC + c];
    int   cn = cnt[b * ROWS + q];
    if (q < NFG) {
        int d = (cn > 0) ? cn : 1;                     // one-hot fallback count = 1
        out[((size_t)b * NFG + q) * CC + c] = s / (float)d;
    } else {
        float r = (cn > 0) ? s / (float)cn : 0.0f;     // bg fallback = 0
        out[(size_t)BB * NFG * CC + ((size_t)b * NBG + (q - NFG)) * CC + c] = r;
    }
}

// ---------------------------------------------------------------------------
extern "C" void kernel_launch(void* const* d_in, const int* in_sizes, int n_in,
                              void* d_out, int out_size, void* d_ws, size_t ws_size,
                              hipStream_t stream) {
    const float* fmap = (const float*)d_in[0];   // [B, C, H, W] f32
    const float* fg   = (const float*)d_in[1];   // [B, I, H, W] f32
    const float* bg   = (const float*)d_in[2];   // [B, Qb, H, W] f32
    float* out = (float*)d_out;                  // fg [B,I,C] ++ bg [B,Qb,C]

    // Workspace partition (needs ~8.6 MiB):
    //  acc : B*32*C f32        (256 KiB)
    //  cnt : B*32 int
    //  amax: B*32 int
    //  W   : B*32*N f16        (8 MiB), 16B-aligned
    float*    acc  = (float*)d_ws;
    int*      cnt  = (int*)(acc + (size_t)BB * ROWS * CC);
    int*      amax = cnt + BB * ROWS;
    _Float16* W    = (_Float16*)(amax + BB * ROWS);

    mask_stats_kernel<<<BB * NQ, 256, 0, stream>>>(fg, bg, cnt, amax);

    {
        size_t total = (size_t)BB * ROWS * NSP;
        int blocks = (int)((total + 255) / 256);
        build_w_kernel<<<blocks, 256, 0, stream>>>(fg, bg, cnt, amax, W);
    }

    zero_acc_kernel<<<(BB * ROWS * CC + 255) / 256, 256, 0, stream>>>(acc);

    wmma_gemm_kernel<<<BB * 16 * KCH, 32, 0, stream>>>(fmap, W, acc);

    finalize_kernel<<<(BB * NQ * CC + 255) / 256, 256, 0, stream>>>(acc, cnt, out);
}
